// gRNN_84146999263835
// MI455X (gfx1250) — matrix operand order repaired and checked by reading, not compiled, not run on hardware
//
#include <hip/hip_runtime.h>

// gRNN over complete binary trees, one tree per workgroup, fully LDS-resident.
// GEMM per level: comb[n_d x 100] @ Ww^T[100 x 100] via v_wmma_f32_16x16x32_f16,
// K padded 100->128 (4 wmma steps), N padded 100->112 (7 col tiles).

typedef _Float16 v16h __attribute__((ext_vector_type(16)));
typedef _Float16 v8h  __attribute__((ext_vector_type(8)));
typedef float    v8f  __attribute__((ext_vector_type(8)));

#define S_DIM   100
#define STR     136                 // padded row stride in halves (272B, bank-staggered, 16B aligned)
#define NODES   1023
#define LEAVES  512
#define H_HALVES (512 * STR)        // h state, leaf level is biggest
#define C_HALVES (256 * STR)        // child-sum (comb) buffer
#define W_HALVES (112 * STR)        // Ww in f16, rows padded to 112, cols to 136 (zeros)
#define F_COUNT  (300 + NODES)      // qw, qb, wb (100 each) + per-tree times
#define SMEM_BYTES ((H_HALVES + C_HALVES + W_HALVES) * 2 + F_COUNT * 4)

__global__ __launch_bounds__(256, 1)
void grnn_tree_kernel(const float* __restrict__ times,
                      const float* __restrict__ Qw, const float* __restrict__ Qb,
                      const float* __restrict__ Ww, const float* __restrict__ Wb,
                      const float* __restrict__ Pw, const float* __restrict__ Pb,
                      float* __restrict__ out)
{
    extern __shared__ _Float16 smem[];
    _Float16* hbuf = smem;                    // [512][STR] current/child level activations (f16)
    _Float16* cbuf = hbuf + H_HALVES;         // [256][STR] comb = h_L + h_R
    _Float16* wbuf = cbuf + C_HALVES;         // [112][STR] Ww row-major f16, zero padded
    float* fbase = (float*)(wbuf + W_HALVES);
    float* qw_s = fbase;
    float* qb_s = fbase + 100;
    float* wb_s = fbase + 200;
    float* t_s  = fbase + 300;                // [1023] this tree's node times

    const int tid  = threadIdx.x;
    const int tree = blockIdx.x;
    const float* tptr = times + (size_t)tree * NODES;

    // ---- init: stage Ww (f16, zero-padded), params, and this tree's times ----
    for (int idx = tid; idx < W_HALVES; idx += 256) {
        int r = idx / STR, c = idx % STR;
        float v = (r < S_DIM && c < S_DIM) ? Ww[r * S_DIM + c] : 0.0f;
        wbuf[idx] = (_Float16)v;
    }
    if (tid < S_DIM) {
        qw_s[tid] = Qw[tid];
        qb_s[tid] = Qb[tid];
        wb_s[tid] = Wb[tid];
    }
    for (int i = tid; i < NODES; i += 256) t_s[i] = tptr[i];
    __syncthreads();

    // ---- leaf level: h[512][c] = relu(t*Qw + Qb), pad columns zeroed ----
    for (int idx = tid; idx < H_HALVES; idx += 256) {
        int r = idx / STR, c = idx % STR;
        float v = 0.0f;
        if (c < S_DIM) {
            float t = t_s[(LEAVES - 1) + r];
            v = t * qw_s[c] + qb_s[c];
            v = v > 0.0f ? v : 0.0f;
        }
        hbuf[idx] = (_Float16)v;
    }
    __syncthreads();

    const int wave  = tid >> 5;
    const int lane  = tid & 31;
    const int row16 = lane & 15;           // A row / B col / D col within a 16-tile
    const int ahalf = (lane >> 4) * 8;     // A K sub-block base (halves)
    const int bhalf = (lane >> 4) * 8;     // B K sub-block base (halves)

    // ---- level loop: 256 -> 1 internal nodes ----
    for (int nd = 256; nd >= 1; nd >>= 1) {
        const int s_off    = nd - 1;                 // heap offset of this level
        const int rows_pad = (nd < 16) ? 16 : nd;    // pad M to a full 16-tile

        // comb[i] = h[2i] + h[2i+1], zero rows beyond nd (pad cols already zero in h)
        for (int idx = tid; idx < rows_pad * STR; idx += 256) {
            int r = idx / STR, c = idx % STR;
            _Float16 v = (_Float16)0.0f;
            if (r < nd) v = hbuf[(2 * r) * STR + c] + hbuf[(2 * r + 1) * STR + c];
            cbuf[idx] = v;
        }
        __syncthreads();

        const int mtiles = rows_pad >> 4;
        const int tiles  = mtiles * 7;               // 7 N-tiles cover 112 >= 100 cols
        for (int tile = wave; tile < tiles; tile += 8) {   // wave-uniform -> EXEC all-ones
            const int mt = tile / 7, nt = tile % 7;
            const _Float16* ap = cbuf + (mt * 16 + row16) * STR + ahalf;
            const _Float16* bp = wbuf + (nt * 16 + row16) * STR + bhalf;
            v8f acc = {};
            #pragma unroll
            for (int kc = 0; kc < 4; ++kc) {
                const int k0 = kc * 32;
                // A 16x32 f16 layout: halves 0-7 -> K=kb+0..7, halves 8-15 -> K=kb+16..23
                v8h alo = *(const v8h*)(ap + k0);
                v8h ahi = *(const v8h*)(ap + k0 + 16);
                // B 32x16 f16 layout: halves 0-7 -> K=kb+0..7, halves 8-15 -> K=kb+16..23
                v8h blo = *(const v8h*)(bp + k0);
                v8h bhi = *(const v8h*)(bp + k0 + 16);
                v16h a = __builtin_shufflevector(alo, ahi, 0,1,2,3,4,5,6,7,8,9,10,11,12,13,14,15);
                v16h b = __builtin_shufflevector(blo, bhi, 0,1,2,3,4,5,6,7,8,9,10,11,12,13,14,15);
                acc = __builtin_amdgcn_wmma_f32_16x16x32_f16(
                        /*neg_a=*/false, a, /*neg_b=*/false, b,
                        /*c_mod=*/(short)0, acc, /*reuse_a=*/false, /*reuse_b=*/false);
            }
            // fused epilogue: + Wb + t*Qw + Qb, relu, f16 store into next-level h
            const int n = nt * 16 + row16;
            #pragma unroll
            for (int r = 0; r < 8; ++r) {
                const int m = mt * 16 + (lane >> 4) * 8 + r;   // D row per C/D layout
                if (n < S_DIM) {
                    float t = t_s[s_off + m];
                    float v = acc[r] + wb_s[n] + t * qw_s[n] + qb_s[n];
                    v = v > 0.0f ? v : 0.0f;
                    hbuf[m * STR + n] = (_Float16)v;
                }
            }
        }
        __syncthreads();
    }

    // ---- root projection: out[p] = root . Pw[p] + Pb[p], p < 5 ----
    if (tid < 5) {
        float acc = Pb[tid];
        for (int s = 0; s < S_DIM; ++s)
            acc += (float)hbuf[s] * Pw[tid * S_DIM + s];
        out[tree * 5 + tid] = acc;
    }
}

extern "C" void kernel_launch(void* const* d_in, const int* in_sizes, int n_in,
                              void* d_out, int out_size, void* d_ws, size_t ws_size,
                              hipStream_t stream) {
    const float* times = (const float*)d_in[0];
    const float* Qw    = (const float*)d_in[1];
    const float* Qb    = (const float*)d_in[2];
    const float* Ww    = (const float*)d_in[3];
    const float* Wb    = (const float*)d_in[4];
    const float* Pw    = (const float*)d_in[5];
    const float* Pb    = (const float*)d_in[6];
    float* out = (float*)d_out;

    const int n_trees = in_sizes[0] / NODES;   // 1024

    // Opt into >64KB dynamic LDS (CDNA5 WGP has 320KB); deterministic, capture-safe.
    hipFuncSetAttribute((const void*)grnn_tree_kernel,
                        hipFuncAttributeMaxDynamicSharedMemorySize, (int)SMEM_BYTES);

    grnn_tree_kernel<<<n_trees, 256, SMEM_BYTES, stream>>>(
        times, Qw, Qb, Ww, Wb, Pw, Pb, out);
}
